// HypoAttention_19258633355493
// MI455X (gfx1250) — matrix operand order repaired and checked by reading, not compiled, not run on hardware
//
#include <hip/hip_runtime.h>
#include <math.h>

// ---------------------------------------------------------------------------
// MI455X (gfx1250) fused HypoAttention:
//   ~138 GFLOP vs ~70MB traffic (~2000 FLOP/byte) => compute bound.
//   All matmuls on v_wmma_f32_16x16x32_bf16; GEMM tiles staged by the
//   Tensor Data Mover (tensor_load_to_lds + s_wait_tensorcnt, double buffered).
// ---------------------------------------------------------------------------

typedef __bf16 bf16_t;
typedef __attribute__((ext_vector_type(16))) __bf16 v16bf;
typedef __attribute__((ext_vector_type(8)))  __bf16 v8bf;
typedef __attribute__((ext_vector_type(8)))  float  v8f;
typedef __attribute__((ext_vector_type(4)))  unsigned int u32x4;
typedef __attribute__((ext_vector_type(8)))  int i32x8;
typedef __attribute__((ext_vector_type(4)))  int i32x4;

union ABFrag { v16bf v; v8bf h[2]; };

#define TOK   8192      // B*N
#define NCTX  2048
#define CDIM  1024
#define HDIM  128
#define QKVN  3072
#define TTDIM 1536

// ---------------------------------------------------------------------------
// TDM: DMA one 128-row x 32-bf16 (16 DWORD) tile, row stride 512 DWORDs,
// into LDS with 4-DWORD padding every 16 DWORDs (=> 40-half LDS row pitch).
// D# per CDNA5 ISA sec.8. Tensor dims huge => no OOB clip.
// ---------------------------------------------------------------------------
__device__ __forceinline__ void tdm_load_tile(unsigned lds_off, const bf16_t* gptr) {
  unsigned long long ga = (unsigned long long)(uintptr_t)gptr;
  u32x4 g0;
  g0[0] = 1u;                                              // count=1, user mode
  g0[1] = lds_off;                                         // LDS byte address
  g0[2] = (unsigned)(ga & 0xffffffffull);                  // global_addr[31:0]
  g0[3] = (unsigned)((ga >> 32) & 0x01ffffffull) | 0x80000000u;  // [56:32] | type=2
  i32x8 g1;
  g1[0] = (int)0x06D20000;   // data_size=4B | pad_enable | pad_interval=16DW | pad_amount=4DW
  g1[1] = 0;                 // no atomic barrier; tensor_dim0[15:0]=0
  g1[2] = 0x00000010;        // tensor_dim0=2^20 DW; tensor_dim1[15:0]=0
  g1[3] = 0x00100010;        // tensor_dim1=2^20; tile_dim0=16 DW
  g1[4] = 128;               // tile_dim1=128 rows, tile_dim2=0
  g1[5] = 512;               // tensor_dim0_stride=512 DW (1024 bf16)
  g1[6] = 0;
  g1[7] = 0;
  i32x4 z4 = (i32x4)0;
#if __clang_major__ >= 23
  i32x8 z8 = (i32x8)0;
  __builtin_amdgcn_tensor_load_to_lds(g0, g1, z4, z4, z8, 0);
#else
  __builtin_amdgcn_tensor_load_to_lds(g0, g1, z4, z4, 0);
#endif
}

// ---------------------------------------------------------------------------
// W_tt[t, c] = sum_{b,cc} core1[0,i,p,b] core2[b,j,q,cc] core3[cc,k,r,0]
// ---------------------------------------------------------------------------
__global__ void wtt_kernel(const float* __restrict__ c1, const float* __restrict__ c2,
                           const float* __restrict__ c3, float* __restrict__ wtt) {
  int idx = blockIdx.x * blockDim.x + threadIdx.x;
  if (idx >= TTDIM * CDIM) return;
  int t = idx >> 10;
  int c = idx & 1023;
  int i = t / 192, j = (t >> 4) % 12, k = t & 15;
  int p = c >> 7, q = (c >> 4) & 7, r = c & 15;
  float acc = 0.f;
  for (int b = 0; b < 8; ++b) {
    float a = c1[(i * 8 + p) * 8 + b];
    for (int cc = 0; cc < 8; ++cc) {
      acc += a * c2[((b * 12 + j) * 8 + q) * 8 + cc] * c3[(cc * 16 + k) * 16 + r];
    }
  }
  wtt[idx] = acc;
}

// Pack QKV weight into per-head rows (heads 0..3 dense, 4..7 tensor-train).
__global__ void pack_qkv_kernel(const float* __restrict__ wd, const float* __restrict__ bd,
                                const float* __restrict__ wtt, const float* __restrict__ btt,
                                bf16_t* __restrict__ wqkv, float* __restrict__ bqkv) {
  int idx = blockIdx.x * blockDim.x + threadIdx.x;
  if (idx >= QKVN * CDIM) return;
  int f = idx >> 10, c = idx & 1023;
  int part = f >> 10, rem = f & 1023;
  int h = rem >> 7, d = rem & 127;
  float w, bv;
  if (h < 4) { int s = part * 512 + h * 128 + d;       w = wd[s * CDIM + c];  bv = bd[s]; }
  else       { int s = part * 512 + (h - 4) * 128 + d; w = wtt[s * CDIM + c]; bv = btt[s]; }
  wqkv[idx] = (bf16_t)w;
  if (c == 0) bqkv[f] = bv;
}

__global__ void f32_to_bf16_kernel(const float* __restrict__ src, bf16_t* __restrict__ dst, int n) {
  int i = blockIdx.x * blockDim.x + threadIdx.x;
  if (i < n) dst[i] = (bf16_t)src[i];
}

// ---------------------------------------------------------------------------
// QKV GEMM with TDM double-buffered staging. Block tile 128Mx128N, 8 waves.
// ---------------------------------------------------------------------------
__global__ __launch_bounds__(256) void gemm_qkv_kernel(
    const bf16_t* __restrict__ X, const bf16_t* __restrict__ W,
    const float* __restrict__ bias, bf16_t* __restrict__ QKV) {
  __shared__ bf16_t As[2][128][40];
  __shared__ bf16_t Bs[2][128][40];
  const int tid = threadIdx.x;
  const int wave = tid >> 5, lane = tid & 31;
  const int half = lane >> 4, l16 = lane & 15;
  const int m0 = blockIdx.x * 128;
  const int n0 = blockIdx.y * 128;
  const bf16_t* Ab = X + (size_t)m0 * CDIM;
  const bf16_t* Bb = W + (size_t)n0 * CDIM;

  v8f acc[8];
  for (int t = 0; t < 8; ++t) acc[t] = (v8f)0.f;

  if (wave == 0) {
    tdm_load_tile((unsigned)(uintptr_t)&As[0][0][0], Ab);
    tdm_load_tile((unsigned)(uintptr_t)&Bs[0][0][0], Bb);
  }
  int cur = 0;
  for (int k0 = 0; k0 < CDIM; k0 += 32) {
    if (wave == 0) {
      if (k0 + 32 < CDIM) {
        tdm_load_tile((unsigned)(uintptr_t)&As[cur ^ 1][0][0], Ab + k0 + 32);
        tdm_load_tile((unsigned)(uintptr_t)&Bs[cur ^ 1][0][0], Bb + k0 + 32);
        __builtin_amdgcn_s_wait_tensorcnt(2);  // current buffer's 2 DMAs done
      } else {
        __builtin_amdgcn_s_wait_tensorcnt(0);
      }
    }
    __syncthreads();
    ABFrag a;
    const bf16_t* arow = &As[cur][wave * 16 + l16][0];
    a.h[0] = *(const v8bf*)(arow + 8 * half);
    a.h[1] = *(const v8bf*)(arow + 16 + 8 * half);
    ABFrag bfr[8];
    for (int t = 0; t < 8; ++t) {
      const bf16_t* brow = &Bs[cur][t * 16 + l16][8 * half];
      bfr[t].h[0] = *(const v8bf*)(brow);
      bfr[t].h[1] = *(const v8bf*)(brow + 16);
    }
    for (int t = 0; t < 8; ++t)
      acc[t] = __builtin_amdgcn_wmma_f32_16x16x32_bf16(false, a.v, false, bfr[t].v,
                                                       (short)0, acc[t], false, false);
    __syncthreads();
    cur ^= 1;
  }

  for (int t = 0; t < 8; ++t) {
    int f = n0 + t * 16 + l16;
    float bv = bias[f];
    int part = f >> 10, rem = f & 1023;
    int h = rem >> 7, d = rem & 127;
    for (int j = 0; j < 8; ++j) {
      int tok = m0 + wave * 16 + j + 8 * half;
      int bb = tok >> 11, n = tok & 2047;
      size_t o = ((size_t)(((part * 4 + bb) * 8 + h) * 2048 + n) << 7) + d;
      QKV[o] = (bf16_t)(acc[t][j] + bv);
    }
  }
}

// ---------------------------------------------------------------------------
// Flash attention: 128 queries per block per (b,h); 64-key tiles.
// ---------------------------------------------------------------------------
__global__ __launch_bounds__(256) void attn_kernel(
    const bf16_t* __restrict__ QKV, bf16_t* __restrict__ AO) {
  __shared__ bf16_t Qs[128][136];
  __shared__ bf16_t Ks[64][136];
  __shared__ bf16_t Vt[128][72];   // transposed [d][key]
  __shared__ bf16_t Ps[128][72];   // [q][key]
  const int tid = threadIdx.x;
  const int wave = tid >> 5, lane = tid & 31;
  const int half = lane >> 4, l16 = lane & 15;
  const int q0 = blockIdx.x * 128;
  const int bh = blockIdx.y;
  const size_t part = (size_t)32 * NCTX * HDIM;
  const bf16_t* Qp = QKV + (size_t)bh * NCTX * HDIM;
  const bf16_t* Kp = Qp + part;
  const bf16_t* Vp = Qp + 2 * part;
  const float scale = 0.08838834764831845f;

  for (int r = 0; r < 8; ++r) {
    int lin = (r * 256 + tid) * 8;
    int row = lin >> 7, col = lin & 127;
    *(v8bf*)&Qs[row][col] = *(const v8bf*)&Qp[(size_t)(q0 + row) * HDIM + col];
  }

  float rmax[8], rsum[8];
  v8f oacc[8];
  for (int j = 0; j < 8; ++j) { rmax[j] = -1e30f; rsum[j] = 0.f; }
  for (int dt = 0; dt < 8; ++dt) oacc[dt] = (v8f)0.f;

  for (int kt = 0; kt < NCTX; kt += 64) {
    __syncthreads();
    for (int r = 0; r < 2; ++r) {
      int lin = (r * 256 + tid) * 8;
      int row = lin >> 7, col = lin & 127;
      *(v8bf*)&Ks[row][col] = *(const v8bf*)&Kp[(size_t)(kt + row) * HDIM + col];
      v8bf vv = *(const v8bf*)&Vp[(size_t)(kt + row) * HDIM + col];
      for (int i = 0; i < 8; ++i) Vt[col + i][row] = vv[i];
    }
    __syncthreads();

    v8f sacc[4];
    for (int t = 0; t < 4; ++t) sacc[t] = (v8f)0.f;
    for (int kd = 0; kd < 128; kd += 32) {
      ABFrag a;
      const bf16_t* arow = &Qs[wave * 16 + l16][kd];
      a.h[0] = *(const v8bf*)(arow + 8 * half);
      a.h[1] = *(const v8bf*)(arow + 16 + 8 * half);
      ABFrag bfr[4];
      for (int t = 0; t < 4; ++t) {
        const bf16_t* brow = &Ks[t * 16 + l16][kd + 8 * half];
        bfr[t].h[0] = *(const v8bf*)brow;
        bfr[t].h[1] = *(const v8bf*)(brow + 16);
      }
      for (int t = 0; t < 4; ++t)
        sacc[t] = __builtin_amdgcn_wmma_f32_16x16x32_bf16(false, a.v, false, bfr[t].v,
                                                          (short)0, sacc[t], false, false);
    }

    float p[4][8];
    for (int j = 0; j < 8; ++j) {
      float mx = -1e30f;
      for (int t = 0; t < 4; ++t) { float s = sacc[t][j] * scale; p[t][j] = s; mx = fmaxf(mx, s); }
      for (int m = 1; m < 16; m <<= 1) mx = fmaxf(mx, __shfl_xor(mx, m, 32));
      float mnew = fmaxf(rmax[j], mx);
      float alpha = __expf(rmax[j] - mnew);
      float rs = 0.f;
      for (int t = 0; t < 4; ++t) { float e = __expf(p[t][j] - mnew); p[t][j] = e; rs += e; }
      for (int m = 1; m < 16; m <<= 1) rs += __shfl_xor(rs, m, 32);
      rsum[j] = rsum[j] * alpha + rs;
      rmax[j] = mnew;
      for (int dt = 0; dt < 8; ++dt) oacc[dt][j] *= alpha;
    }

    for (int t = 0; t < 4; ++t)
      for (int j = 0; j < 8; ++j)
        Ps[wave * 16 + j + 8 * half][t * 16 + l16] = (bf16_t)p[t][j];
    __syncthreads();

    for (int ks = 0; ks < 64; ks += 32) {
      ABFrag a;
      const bf16_t* arow = &Ps[wave * 16 + l16][ks];
      a.h[0] = *(const v8bf*)(arow + 8 * half);
      a.h[1] = *(const v8bf*)(arow + 16 + 8 * half);
      ABFrag bfr[8];
      for (int dt = 0; dt < 8; ++dt) {
        const bf16_t* brow = &Vt[dt * 16 + l16][ks + 8 * half];
        bfr[dt].h[0] = *(const v8bf*)brow;
        bfr[dt].h[1] = *(const v8bf*)(brow + 16);
      }
      for (int dt = 0; dt < 8; ++dt)
        oacc[dt] = __builtin_amdgcn_wmma_f32_16x16x32_bf16(false, a.v, false, bfr[dt].v,
                                                           (short)0, oacc[dt], false, false);
    }
  }

  const int b = bh >> 3, h = bh & 7;
  for (int dt = 0; dt < 8; ++dt) {
    for (int j = 0; j < 8; ++j) {
      int qrow = q0 + wave * 16 + j + 8 * half;
      size_t tok = (size_t)b * NCTX + qrow;
      int col = h * HDIM + dt * 16 + l16;
      AO[tok * CDIM + col] = (bf16_t)(oacc[dt][j] / rsum[j]);
    }
  }
}

// ---------------------------------------------------------------------------
// Output projection with TDM staging: AO(8192x1024) @ w_out^T + b_out -> f32
// ---------------------------------------------------------------------------
__global__ __launch_bounds__(256) void gemm_out_kernel(
    const bf16_t* __restrict__ A, const bf16_t* __restrict__ W,
    const float* __restrict__ bias, float* __restrict__ out) {
  __shared__ bf16_t As[2][128][40];
  __shared__ bf16_t Bs[2][128][40];
  const int tid = threadIdx.x;
  const int wave = tid >> 5, lane = tid & 31;
  const int half = lane >> 4, l16 = lane & 15;
  const int m0 = blockIdx.x * 128;
  const int n0 = blockIdx.y * 128;
  const bf16_t* Ab = A + (size_t)m0 * CDIM;
  const bf16_t* Bb = W + (size_t)n0 * CDIM;

  v8f acc[8];
  for (int t = 0; t < 8; ++t) acc[t] = (v8f)0.f;

  if (wave == 0) {
    tdm_load_tile((unsigned)(uintptr_t)&As[0][0][0], Ab);
    tdm_load_tile((unsigned)(uintptr_t)&Bs[0][0][0], Bb);
  }
  int cur = 0;
  for (int k0 = 0; k0 < CDIM; k0 += 32) {
    if (wave == 0) {
      if (k0 + 32 < CDIM) {
        tdm_load_tile((unsigned)(uintptr_t)&As[cur ^ 1][0][0], Ab + k0 + 32);
        tdm_load_tile((unsigned)(uintptr_t)&Bs[cur ^ 1][0][0], Bb + k0 + 32);
        __builtin_amdgcn_s_wait_tensorcnt(2);
      } else {
        __builtin_amdgcn_s_wait_tensorcnt(0);
      }
    }
    __syncthreads();
    ABFrag a;
    const bf16_t* arow = &As[cur][wave * 16 + l16][0];
    a.h[0] = *(const v8bf*)(arow + 8 * half);
    a.h[1] = *(const v8bf*)(arow + 16 + 8 * half);
    ABFrag bfr[8];
    for (int t = 0; t < 8; ++t) {
      const bf16_t* brow = &Bs[cur][t * 16 + l16][8 * half];
      bfr[t].h[0] = *(const v8bf*)(brow);
      bfr[t].h[1] = *(const v8bf*)(brow + 16);
    }
    for (int t = 0; t < 8; ++t)
      acc[t] = __builtin_amdgcn_wmma_f32_16x16x32_bf16(false, a.v, false, bfr[t].v,
                                                       (short)0, acc[t], false, false);
    __syncthreads();
    cur ^= 1;
  }

  for (int t = 0; t < 8; ++t) {
    int f = n0 + t * 16 + l16;
    float bv = bias[f];
    for (int j = 0; j < 8; ++j) {
      int tok = m0 + wave * 16 + j + 8 * half;
      out[(size_t)tok * CDIM + f] = acc[t][j] + bv;
    }
  }
}

// ---------------------------------------------------------------------------
extern "C" void kernel_launch(void* const* d_in, const int* in_sizes, int n_in,
                              void* d_out, int out_size, void* d_ws, size_t ws_size,
                              hipStream_t stream) {
  const float* x   = (const float*)d_in[0];
  const float* wd  = (const float*)d_in[1];
  const float* bd  = (const float*)d_in[2];
  const float* c1  = (const float*)d_in[3];
  const float* c2  = (const float*)d_in[4];
  const float* c3  = (const float*)d_in[5];
  const float* btt = (const float*)d_in[6];
  const float* wo  = (const float*)d_in[7];
  const float* bo  = (const float*)d_in[8];

  char* ws = (char*)d_ws;
  size_t off = 0;
  auto alloc = [&](size_t bytes) -> char* {
    char* p = ws + off;
    off += (bytes + 255) & ~(size_t)255;
    return p;
  };
  float*  Wtt  = (float*) alloc((size_t)TTDIM * CDIM * 4);
  bf16_t* Wqkv = (bf16_t*)alloc((size_t)QKVN * CDIM * 2);
  float*  bqkv = (float*) alloc((size_t)QKVN * 4);
  bf16_t* Wob  = (bf16_t*)alloc((size_t)CDIM * CDIM * 2);
  bf16_t* Xb   = (bf16_t*)alloc((size_t)TOK * CDIM * 2);
  bf16_t* QKVb = (bf16_t*)alloc((size_t)3 * 32 * NCTX * HDIM * 2);
  bf16_t* AOb  = (bf16_t*)alloc((size_t)TOK * CDIM * 2);

  wtt_kernel<<<(TTDIM * CDIM + 255) / 256, 256, 0, stream>>>(c1, c2, c3, Wtt);
  pack_qkv_kernel<<<(QKVN * CDIM + 255) / 256, 256, 0, stream>>>(wd, bd, Wtt, btt, Wqkv, bqkv);
  f32_to_bf16_kernel<<<(TOK * CDIM + 255) / 256, 256, 0, stream>>>(x, Xb, TOK * CDIM);
  f32_to_bf16_kernel<<<(CDIM * CDIM + 255) / 256, 256, 0, stream>>>(wo, Wob, CDIM * CDIM);

  gemm_qkv_kernel<<<dim3(TOK / 128, QKVN / 128), 256, 0, stream>>>(Xb, Wqkv, bqkv, QKVb);
  attn_kernel<<<dim3(NCTX / 128, 32), 256, 0, stream>>>(QKVb, AOb);
  gemm_out_kernel<<<dim3(TOK / 128, CDIM / 128), 256, 0, stream>>>(AOb, Wob, bo, (float*)d_out);
}
